// HolographicAttention_58987080843590
// MI455X (gfx1250) — matrix operand order repaired and checked by reading, not compiled, not run on hardware
//
#include <hip/hip_runtime.h>
#include <math.h>
#include <stdint.h>

// ---------- types ----------
typedef __attribute__((ext_vector_type(16))) __bf16 v16bf;
typedef __attribute__((ext_vector_type(8)))  __bf16 v8bf;
typedef __attribute__((ext_vector_type(8)))  float  v8f;
typedef __attribute__((ext_vector_type(4)))  unsigned int u32x4;
typedef __attribute__((ext_vector_type(8)))  int i32x8;
typedef __attribute__((ext_vector_type(4)))  int i32x4;
using as1_v4i = __attribute__((address_space(1))) i32x4;
using as3_v4i = __attribute__((address_space(3))) i32x4;

#if __has_builtin(__builtin_amdgcn_global_load_async_to_lds_b128)
#define ASYNC_LDS_OK 1
#else
#define ASYNC_LDS_OK 0
#endif
#if __has_builtin(__builtin_amdgcn_tensor_load_to_lds)
#define TDM_OK 1
#else
#define TDM_OK 0
#endif

static __device__ inline void wait_async_zero() {
#if __has_builtin(__builtin_amdgcn_s_wait_asynccnt)
  __builtin_amdgcn_s_wait_asynccnt(0);
#else
  asm volatile("s_wait_asynccnt 0" ::: "memory");
#endif
}

#define B_  2
#define S_  2048
#define D_  2048
#define H_  16
#define DH_ 128

static __device__ inline v16bf mk16(v8bf lo, v8bf hi) {
  v16bf r;
#pragma unroll
  for (int i = 0; i < 8; ++i) { r[i] = lo[i]; r[8 + i] = hi[i]; }
  return r;
}

// ================= 0) f32 -> bf16 convert =================
__global__ void cvt_bf16_kernel(const float* __restrict__ src, __bf16* __restrict__ dst, int n) {
  int i = blockIdx.x * blockDim.x + threadIdx.x;
  if (i < n) dst[i] = (__bf16)src[i];
}

// ================= 1) bf16 GEMM:  C[m,n] = sum_k A[m,k]*B[n,k] =================
// A: MxK row-major bf16, B: NxK row-major bf16 (C = A @ B^T), C: MxN f32.
// Block tile 64x128, 8 waves (2x4), each wave 32x32 (2x2 WMMA tiles), K-step 32.
// A tile staged by the Tensor Data Mover (TDM) with pad_enable reproducing the
// padded LDS row stride; B tile staged by per-lane async global->LDS copies.
#define GBM 64
#define GBN 128
#define GKB 32
#define LDR 40   // LDS row stride in halves (32 + 8 pad) = 80B, 16B aligned

__global__ __launch_bounds__(256) void gemm_bf16_nt(const __bf16* __restrict__ A,
                                                    const __bf16* __restrict__ B,
                                                    float* __restrict__ C,
                                                    int M, int N, int K) {
  __shared__ __bf16 As[GBM * LDR];
  __shared__ __bf16 Bs[GBN * LDR];

  const int tid  = threadIdx.x;
  const int lane = tid & 31;
  const int wave = tid >> 5;
  const int lq   = lane & 15;
  const int hi   = lane >> 4;
  const int wm   = wave >> 2;   // 0..1
  const int wn   = wave & 3;    // 0..3
  const int bm0  = blockIdx.y * GBM;
  const int bn0  = blockIdx.x * GBN;

  v8f acc[2][2];
#pragma unroll
  for (int i = 0; i < 2; ++i)
#pragma unroll
    for (int j = 0; j < 2; ++j) { v8f z = {}; acc[i][j] = z; }

  for (int k0 = 0; k0 < K; k0 += GKB) {
    // ---- stage A tile (64 rows x 32 halves) via TDM ----
#if TDM_OK
    if (tid == 0) {
      unsigned long long ga = (unsigned long long)(uintptr_t)(A + (size_t)bm0 * K + k0);
      unsigned int ldsa = (unsigned int)(uintptr_t)(&As[0]);
      u32x4 g0;
      g0[0] = 1u;                                        // count=1 valid descriptor
      g0[1] = ldsa;                                      // lds_addr (bytes)
      g0[2] = (unsigned int)ga;                          // global_addr[31:0]
      g0[3] = (unsigned int)(ga >> 32) | (2u << 30);     // global_addr[56:32] | type=2
      i32x8 g1;
      // data_size=2B (code 1), pad_enable, pad_interval=16 DW (code 3),
      // pad_amount=4 DW (code 3)  ->  row 64B + 16B pad = 80B = LDR halves
      g1[0] = (int)((1u << 16) | (1u << 20) | (3u << 22) | (3u << 25));
      g1[1] = (int)(((unsigned)K & 0xFFFFu) << 16);      // tensor_dim0[15:0]
      g1[2] = (int)(((unsigned)K >> 16) & 0xFFFFu);      // tensor_dim0[31:16], dim1 lo=0
      g1[3] = (int)(0x10u | (32u << 16));                // tensor_dim1=1<<20, tile_dim0=32
      g1[4] = 64;                                        // tile_dim1=64 rows
      g1[5] = K;                                         // tensor_dim0_stride[31:0]
      g1[6] = 0;
      g1[7] = 0;
      i32x4 z4 = {0, 0, 0, 0};
#if __clang_major__ >= 23
      i32x8 z8 = {0, 0, 0, 0, 0, 0, 0, 0};
      __builtin_amdgcn_tensor_load_to_lds(g0, g1, z4, z4, z8, 0);
#else
      __builtin_amdgcn_tensor_load_to_lds(g0, g1, z4, z4, 0);
#endif
      __builtin_amdgcn_s_wait_tensorcnt(0);
    }
#else
    {
      const int arow = tid >> 2, ach = tid & 3;          // 64 rows x 4 chunks of 16B
      const __bf16* gp = A + (size_t)(bm0 + arow) * K + k0 + ach * 8;
      v8bf d = *(const v8bf*)gp;
      *(v8bf*)(&As[arow * LDR + ach * 8]) = d;
    }
#endif
    // ---- stage B tile (128 rows x 32 halves) via async global->LDS ----
#if ASYNC_LDS_OK
#pragma unroll
    for (int j = 0; j < 2; ++j) {
      int idx = tid * 2 + j;                             // 0..511
      int row = idx >> 2, ch = idx & 3;
      const __bf16* gp = B + (size_t)(bn0 + row) * K + k0 + ch * 8;
      as1_v4i* gpp = (as1_v4i*)(uintptr_t)gp;
      as3_v4i* lp  = (as3_v4i*)(unsigned int)(uintptr_t)(&Bs[row * LDR + ch * 8]);
      __builtin_amdgcn_global_load_async_to_lds_b128(gpp, lp, 0, 0);
    }
    wait_async_zero();
#else
#pragma unroll
    for (int j = 0; j < 2; ++j) {
      int idx = tid * 2 + j;
      int row = idx >> 2, ch = idx & 3;
      const __bf16* gp = B + (size_t)(bn0 + row) * K + k0 + ch * 8;
      v8bf d = *(const v8bf*)gp;
      *(v8bf*)(&Bs[row * LDR + ch * 8]) = d;
    }
#endif
    __syncthreads();

    v16bf af[2], bfr[2];
#pragma unroll
    for (int mi = 0; mi < 2; ++mi) {
      int r = wm * 32 + mi * 16 + lq;
      // A frag: halves 0..7 = K hi*8+0..7 ; halves 8..15 = K 16+hi*8+0..7
      v8bf lo = *(const v8bf*)(&As[r * LDR + hi * 8]);
      v8bf hh = *(const v8bf*)(&As[r * LDR + 16 + hi * 8]);
      af[mi] = mk16(lo, hh);
    }
#pragma unroll
    for (int ni = 0; ni < 2; ++ni) {
      int r = wn * 32 + ni * 16 + lq;
      // B frag: halves 0..7 = K hi*8+0..7 ; halves 8..15 = K 16+hi*8+0..7
      v8bf lo = *(const v8bf*)(&Bs[r * LDR + hi * 8]);
      v8bf hh = *(const v8bf*)(&Bs[r * LDR + 16 + hi * 8]);
      bfr[ni] = mk16(lo, hh);
    }
#pragma unroll
    for (int mi = 0; mi < 2; ++mi)
#pragma unroll
      for (int ni = 0; ni < 2; ++ni)
        acc[mi][ni] = __builtin_amdgcn_wmma_f32_16x16x32_bf16(
            false, af[mi], false, bfr[ni], (short)0, acc[mi][ni], false, false);
    __syncthreads();
  }

  // epilogue: C layout -> lane = col (N), vgpr r = row r (+8 for hi lanes)
#pragma unroll
  for (int mi = 0; mi < 2; ++mi)
#pragma unroll
    for (int ni = 0; ni < 2; ++ni) {
      int col = bn0 + wn * 32 + ni * 16 + lq;
#pragma unroll
      for (int r = 0; r < 8; ++r) {
        int row = bm0 + wm * 32 + mi * 16 + r + hi * 8;
        C[(size_t)row * N + col] = acc[mi][ni][r];
      }
    }
}

// ================= 2) RoPE + layout split =================
__global__ void rope_split_kernel(const float* __restrict__ qkv,
                                  const float* __restrict__ cosp,
                                  const float* __restrict__ sinp,
                                  __bf16* __restrict__ qb, __bf16* __restrict__ kb,
                                  __bf16* __restrict__ vt) {
  int idx = blockIdx.x * blockDim.x + threadIdx.x;  // B*H*S*64
  if (idx >= B_ * H_ * S_ * (DH_ / 2)) return;
  int i  = idx & 63;
  int s  = (idx >> 6) & (S_ - 1);
  int bh = idx >> 17;
  int b  = bh >> 4, h = bh & 15;

  size_t m    = (size_t)b * S_ + s;
  size_t base = m * (3 * D_);
  int    col  = h * DH_ + 2 * i;

  float c = cosp[s * 64 + i], sn = sinp[s * 64 + i];
  float q1 = qkv[base + col],            q2 = qkv[base + col + 1];
  float k1 = qkv[base + D_ + col],       k2 = qkv[base + D_ + col + 1];
  float v1 = qkv[base + 2 * D_ + col],   v2 = qkv[base + 2 * D_ + col + 1];

  size_t qo = ((size_t)bh * S_ + s) * DH_ + 2 * i;
  qb[qo]     = (__bf16)(q1 * c - q2 * sn);
  qb[qo + 1] = (__bf16)(q1 * sn + q2 * c);
  kb[qo]     = (__bf16)(k1 * c - k2 * sn);
  kb[qo + 1] = (__bf16)(k1 * sn + k2 * c);
  vt[((size_t)bh * DH_ + 2 * i) * S_ + s]     = (__bf16)v1;
  vt[((size_t)bh * DH_ + 2 * i + 1) * S_ + s] = (__bf16)v2;
}

// ================= 3) flash attention (causal), one wave per 16 queries =================
__global__ __launch_bounds__(128) void flash_attn_kernel(const __bf16* __restrict__ qb,
                                                         const __bf16* __restrict__ kb,
                                                         const __bf16* __restrict__ vt,
                                                         float* __restrict__ aout) {
  const int W    = blockIdx.x * 4 + (threadIdx.x >> 5);
  const int bh   = W >> 7;
  const int qt   = W & 127;
  const int q0   = qt * 16;
  const int lane = threadIdx.x & 31;
  const int lq   = lane & 15;
  const int hi   = lane >> 4;
  const float scl = 0.08838834764831845f;   // 1/sqrt(128)
  const float NEG = -__builtin_inff();

  v16bf qf[4];
  {
    size_t qbase = ((size_t)bh * S_ + (q0 + lq)) * DH_;
#pragma unroll
    for (int t = 0; t < 4; ++t) {
      const __bf16* p = qb + qbase + t * 32 + hi * 8;
      qf[t] = mk16(*(const v8bf*)p, *(const v8bf*)(p + 16));
    }
  }

  v8f o[8];
#pragma unroll
  for (int n = 0; n < 8; ++n) { v8f z = {}; o[n] = z; }
  float m_i = NEG, l_i = 0.0f;
  const int qg = q0 + lq;

  for (int c = 0; c <= q0 + 15; c += 32) {
    v8f s0 = {}, s1 = {};
#pragma unroll
    for (int t = 0; t < 4; ++t) {
      size_t ab = ((size_t)bh * S_ + (c + lq)) * DH_ + t * 32 + hi * 8;
      v16bf a0 = mk16(*(const v8bf*)(kb + ab), *(const v8bf*)(kb + ab + 16));
      s0 = __builtin_amdgcn_wmma_f32_16x16x32_bf16(false, a0, false, qf[t], (short)0, s0, false, false);
      size_t ab1 = ab + (size_t)16 * DH_;
      v16bf a1 = mk16(*(const v8bf*)(kb + ab1), *(const v8bf*)(kb + ab1 + 16));
      s1 = __builtin_amdgcn_wmma_f32_16x16x32_bf16(false, a1, false, qf[t], (short)0, s1, false, false);
    }
    float mblk = NEG;
#pragma unroll
    for (int r = 0; r < 8; ++r) {
      int k0g = c + r + hi * 8;
      float x0 = (k0g <= qg) ? s0[r] * scl : NEG;
      float x1 = (k0g + 16 <= qg) ? s1[r] * scl : NEG;
      s0[r] = x0; s1[r] = x1;
      mblk = fmaxf(mblk, fmaxf(x0, x1));
    }
    mblk = fmaxf(mblk, __shfl_xor(mblk, 16, 32));
    float mnew  = fmaxf(m_i, mblk);
    float alpha = __expf(m_i - mnew);
    v16bf pa;
    float lblk = 0.0f;
#pragma unroll
    for (int r = 0; r < 8; ++r) {
      float p0 = __expf(s0[r] - mnew);
      float p1 = __expf(s1[r] - mnew);
      pa[r]     = (__bf16)p0;
      pa[8 + r] = (__bf16)p1;
      lblk += p0 + p1;
    }
    lblk += __shfl_xor(lblk, 16, 32);
    l_i = l_i * alpha + lblk;
    m_i = mnew;
    float af[8];
#pragma unroll
    for (int r = 0; r < 8; ++r) af[r] = __shfl(alpha, r + hi * 8, 32);
#pragma unroll
    for (int n = 0; n < 8; ++n)
#pragma unroll
      for (int r = 0; r < 8; ++r) o[n][r] *= af[r];
#pragma unroll
    for (int n = 0; n < 8; ++n) {
      size_t vb = ((size_t)bh * DH_ + n * 16 + lq) * S_ + c + hi * 8;
      v16bf vf = mk16(*(const v8bf*)(vt + vb), *(const v8bf*)(vt + vb + 16));
      o[n] = __builtin_amdgcn_wmma_f32_16x16x32_bf16(false, pa, false, vf, (short)0, o[n], false, false);
    }
  }

  float lf[8];
#pragma unroll
  for (int r = 0; r < 8; ++r) lf[r] = 1.0f / __shfl(l_i, r + hi * 8, 32);
  int b = bh >> 4, h = bh & 15;
#pragma unroll
  for (int n = 0; n < 8; ++n) {
    int dh = n * 16 + lq;
#pragma unroll
    for (int r = 0; r < 8; ++r) {
      int q = q0 + r + hi * 8;
      aout[((size_t)(b * S_ + q)) * D_ + h * DH_ + dh] = o[n][r] * lf[r];
    }
  }
}

// ================= 4) holographic part =================
__global__ __launch_bounds__(128) void holo_kernel(const __bf16* __restrict__ qb,
                                                   const __bf16* __restrict__ kb,
                                                   const __bf16* __restrict__ vt,
                                                   float* __restrict__ hout) {
  __shared__ float qs[DH_], ks[DH_], vs[DH_], rs[DH_];
  const int d   = threadIdx.x;
  const int idx = blockIdx.x;          // bh*S + s
  const int bh  = idx >> 11;
  const int s   = idx & (S_ - 1);

  qs[d] = (float)qb[(size_t)idx * DH_ + d];
  ks[d] = (float)kb[(size_t)idx * DH_ + d];
  vs[d] = (float)vt[((size_t)bh * DH_ + d) * S_ + s];
  __syncthreads();

  float r = 0.0f;
#pragma unroll 8
  for (int i = 0; i < DH_; ++i) r += qs[i] * ks[(i + d) & (DH_ - 1)];
  rs[d] = r;
  __syncthreads();

  float out = 0.0f;
#pragma unroll 8
  for (int j = 0; j < DH_; ++j) out += rs[j] * vs[(d - j) & (DH_ - 1)];

  int b = bh >> 4, h = bh & 15;
  hout[((size_t)(b * S_ + s)) * D_ + h * DH_ + d] = out * (1.0f / DH_);
}

// ================= 5) dual LayerNorm + gated blend -> bf16 =================
__global__ __launch_bounds__(256) void blend_ln_kernel(const float* __restrict__ a,
                                                       const float* __restrict__ hh,
                                                       const float* __restrict__ gamma_a,
                                                       const float* __restrict__ beta_a,
                                                       const float* __restrict__ gamma_h,
                                                       const float* __restrict__ beta_h,
                                                       const float* __restrict__ gate,
                                                       __bf16* __restrict__ y) {
  __shared__ float red[256];
  const int row = blockIdx.x;
  const int tid = threadIdx.x;

  float sa = 0, sa2 = 0, sh = 0, sh2 = 0;
  for (int c = tid; c < D_; c += 256) {
    float xa = a[(size_t)row * D_ + c];  sa += xa; sa2 += xa * xa;
    float xh = hh[(size_t)row * D_ + c]; sh += xh; sh2 += xh * xh;
  }
  float vals[4] = {sa, sa2, sh, sh2};
#pragma unroll
  for (int v = 0; v < 4; ++v) {
    red[tid] = vals[v];
    __syncthreads();
    for (int st = 128; st > 0; st >>= 1) {
      if (tid < st) red[tid] += red[tid + st];
      __syncthreads();
    }
    vals[v] = red[0];
    __syncthreads();
  }
  const float inv = 1.0f / (float)D_;
  float ma = vals[0] * inv, va = vals[1] * inv - ma * ma;
  float mh = vals[2] * inv, vh = vals[3] * inv - mh * mh;
  float ra = rsqrtf(va + 1e-5f), rh = rsqrtf(vh + 1e-5f);
  float g = 1.0f / (1.0f + __expf(-gate[0]));
  g = fminf(0.95f, fmaxf(0.05f, g));

  for (int c = tid; c < D_; c += 256) {
    float la = (a[(size_t)row * D_ + c] - ma) * ra * gamma_a[c] + beta_a[c];
    float lh = (hh[(size_t)row * D_ + c] - mh) * rh * gamma_h[c] + beta_h[c];
    y[(size_t)row * D_ + c] = (__bf16)((1.0f - g) * la + g * lh);
  }
}

// ================= launch =================
extern "C" void kernel_launch(void* const* d_in, const int* in_sizes, int n_in,
                              void* d_out, int out_size, void* d_ws, size_t ws_size,
                              hipStream_t stream) {
  const float* x       = (const float*)d_in[0];
  const float* cosp    = (const float*)d_in[1];
  const float* sinp    = (const float*)d_in[2];
  const float* qkv_w   = (const float*)d_in[3];
  const float* out_w   = (const float*)d_in[4];
  const float* gate    = (const float*)d_in[5];
  const float* gamma_a = (const float*)d_in[6];
  const float* beta_a  = (const float*)d_in[7];
  const float* gamma_h = (const float*)d_in[8];
  const float* beta_h  = (const float*)d_in[9];
  float* out = (float*)d_out;

  const size_t M = (size_t)B_ * S_;       // 4096
  char* p = (char*)d_ws;
  __bf16* xb    = (__bf16*)p; p += M * D_ * 2;
  __bf16* wqkvb = (__bf16*)p; p += (size_t)3 * D_ * D_ * 2;
  __bf16* woutb = (__bf16*)p; p += (size_t)D_ * D_ * 2;
  float*  qkvf  = (float*)p;  p += M * 3 * D_ * 4;
  __bf16* q_bf  = (__bf16*)p; p += M * D_ * 2;
  __bf16* k_bf  = (__bf16*)p; p += M * D_ * 2;
  __bf16* v_t   = (__bf16*)p; p += M * D_ * 2;
  float*  a_f   = (float*)p;  p += M * D_ * 4;
  float*  h_f   = (float*)p;  p += M * D_ * 4;
  __bf16* y_bf  = (__bf16*)p; p += M * D_ * 2;

  {
    int n1 = (int)(M * D_);
    cvt_bf16_kernel<<<(n1 + 255) / 256, 256, 0, stream>>>(x, xb, n1);
    int n2 = 3 * D_ * D_;
    cvt_bf16_kernel<<<(n2 + 255) / 256, 256, 0, stream>>>(qkv_w, wqkvb, n2);
    int n3 = D_ * D_;
    cvt_bf16_kernel<<<(n3 + 255) / 256, 256, 0, stream>>>(out_w, woutb, n3);
  }
  {
    dim3 grid((3 * D_) / GBN, (int)(M / GBM));
    gemm_bf16_nt<<<grid, 256, 0, stream>>>(xb, wqkvb, qkvf, (int)M, 3 * D_, D_);
  }
  {
    int n = B_ * H_ * S_ * (DH_ / 2);
    rope_split_kernel<<<(n + 255) / 256, 256, 0, stream>>>(qkvf, cosp, sinp, q_bf, k_bf, v_t);
  }
  flash_attn_kernel<<<(B_ * H_ * (S_ / 16)) / 4, 128, 0, stream>>>(q_bf, k_bf, v_t, a_f);
  holo_kernel<<<B_ * H_ * S_, DH_, 0, stream>>>(q_bf, k_bf, v_t, h_f);
  blend_ln_kernel<<<(int)M, 256, 0, stream>>>(a_f, h_f, gamma_a, beta_a, gamma_h, beta_h, gate, y_bf);
  {
    dim3 grid(D_ / GBN, (int)(M / GBM));
    gemm_bf16_nt<<<grid, 256, 0, stream>>>(y_bf, woutb, out, (int)M, D_, D_);
  }
}
